// AttentionHead_17471926960727
// MI455X (gfx1250) — matrix operand order repaired and checked by reading, not compiled, not run on hardware
//
#include <hip/hip_runtime.h>
#include <hip/hip_bf16.h>
#include <math.h>

#define SEQ    8192
#define EMBED  2048
#define HEAD   128
#define KSPLIT 8
#define KCHUNK (SEQ / KSPLIT)   // 1024 keys per split

typedef __bf16 bf16_t;
typedef bf16_t v16bf __attribute__((ext_vector_type(16)));
typedef float  v8f   __attribute__((ext_vector_type(8)));

union FragU { v16bf v; unsigned int u[8]; uint4 q[2]; };

__device__ __forceinline__ unsigned pack_bf16(float lo, float hi) {
  union { bf16_t h[2]; unsigned u; } t;
  t.h[0] = (bf16_t)lo; t.h[1] = (bf16_t)hi;
  return t.u;
}

__device__ __forceinline__ v8f wmma_bf16(v16bf a, v16bf b, v8f c) {
  return __builtin_amdgcn_wmma_f32_16x16x32_bf16(false, a, false, b, (short)0, c,
                                                 false, false);
}

// 16-bit A-matrix 16x32 (CDNA5 ISA): lane m=lane&15 holds K = {0..7}+8*hi in
// dwords 0..3 and K = {16..23}+8*hi in dwords 4..7 -> two contiguous 16B spans.
__device__ __forceinline__ v16bf load_afrag(const bf16_t* row, int hi) {
  FragU f;
  f.q[0] = *(const uint4*)(row + 8 * hi);
  f.q[1] = *(const uint4*)(row + 16 + 8 * hi);
  return f.v;
}
// 16-bit B-matrix 32x16: lane n=lane&15 holds column n, K = {0..7}+8*hi, {16..23}+8*hi.
__device__ __forceinline__ v16bf load_bfrag(const bf16_t* col, int hi) {
  FragU f;
  f.q[0] = *(const uint4*)(col + 8 * hi);
  f.q[1] = *(const uint4*)(col + 16 + 8 * hi);
  return f.v;
}

// ---------- Phase 0: x -> bf16 (row-major), W{q,k,v} -> bf16 transposed
__global__ __launch_bounds__(256) void pack_kernel(
    const float* __restrict__ x,
    const float* __restrict__ Wq, const float* __restrict__ Wk,
    const float* __restrict__ Wv,
    bf16_t* __restrict__ xb, bf16_t* __restrict__ wtb)
{
  const int b = blockIdx.x;
  if (b < 8192) {                       // x: 16M elems, 8 per thread
    const size_t i0 = ((size_t)b * 256 + threadIdx.x) * 8;
    const float4 a = *(const float4*)(x + i0);
    const float4 c = *(const float4*)(x + i0 + 4);
    uint4 o;
    o.x = pack_bf16(a.x, a.y); o.y = pack_bf16(a.z, a.w);
    o.z = pack_bf16(c.x, c.y); o.w = pack_bf16(c.z, c.w);
    *(uint4*)(xb + i0) = o;
  } else {                              // W transpose: 3 * 128 cols * 256 chunks
    const int t   = (b - 8192) * 256 + threadIdx.x;   // 0..98303
    const int mat = t >> 15;
    const int rem = t & 32767;
    const int col = rem >> 8;           // 0..127
    const int k0  = (rem & 255) * 8;
    const float* W = (mat == 0) ? Wq : (mat == 1) ? Wk : Wv;
    float v[8];
#pragma unroll
    for (int i = 0; i < 8; ++i) v[i] = W[(size_t)(k0 + i) * HEAD + col];
    uint4 o;
    o.x = pack_bf16(v[0], v[1]); o.y = pack_bf16(v[2], v[3]);
    o.z = pack_bf16(v[4], v[5]); o.w = pack_bf16(v[6], v[7]);
    *(uint4*)(wtb + ((size_t)mat * HEAD + col) * EMBED + k0) = o;
  }
}

// ---------- Phase 1: q = (x@Wq+bq)*log2(e)/sqrt(H), k = x@Wk+bk, vT = (x@Wv+bv)^T
__global__ __launch_bounds__(256) void proj_kernel(
    const bf16_t* __restrict__ xb, const bf16_t* __restrict__ wtb,
    const float* __restrict__ bq, const float* __restrict__ bk,
    const float* __restrict__ bv,
    bf16_t* __restrict__ qb, bf16_t* __restrict__ kbuf, bf16_t* __restrict__ vtb)
{
  const int lane = threadIdx.x & 31;
  const int wave = threadIdx.x >> 5;
  const int wid  = blockIdx.x * 8 + wave;        // 0..12287
  const int mat  = wid >> 12;                    // 0=q 1=k 2=v
  const int rem  = wid & 4095;
  const int mtile = rem >> 3;                    // 0..511
  const int ntile = rem & 7;                     // 0..7
  const int n    = lane & 15;
  const int hi   = lane >> 4;
  const int mbase = mtile * 16;
  const int ncol  = ntile * 16 + n;

  const float* bias = (mat == 0) ? bq : (mat == 1) ? bk : bv;
  const bf16_t* xrow = xb + (size_t)(mbase + n) * EMBED;
  const bf16_t* wcol = wtb + ((size_t)mat * HEAD + ncol) * EMBED;

  v8f acc = {};
  for (int kb = 0; kb < EMBED; kb += 32)
    acc = wmma_bf16(load_afrag(xrow + kb, hi), load_bfrag(wcol + kb, hi), acc);

  const float bcol = bias[ncol];
  // q scale = log2(e)/sqrt(128): softmax done in base 2 (p = exp2(S - m))
  const float sc = (mat == 0) ? 0.12751599833553462f : 1.0f;
  bf16_t* dst = (mat == 0) ? qb : (mat == 1) ? kbuf : vtb;
#pragma unroll
  for (int r = 0; r < 8; ++r) {
    const int row = mbase + r + 8 * hi;          // C/D layout: row = r + 8*hi
    const size_t idx = (mat < 2) ? (size_t)row * HEAD + ncol
                                 : (size_t)ncol * SEQ + row;
    dst[idx] = (bf16_t)((acc[r] + bcol) * sc);
  }
}

// ---------- Phase 2: flash attention (split-K), 16 Q rows x 1024 keys / wave
__global__ __launch_bounds__(256) void attn_kernel(
    const bf16_t* __restrict__ q, const bf16_t* __restrict__ k,
    const bf16_t* __restrict__ vT,
    float* __restrict__ pO, float* __restrict__ pm, float* __restrict__ pl)
{
  __shared__ __align__(16) bf16_t pbuf[8][16 * 64];   // per-wave P staging (2KB)
  const int lane = threadIdx.x & 31;
  const int wave = threadIdx.x >> 5;
  const int n    = lane & 15;
  const int hi   = lane >> 4;
  const int wid  = blockIdx.x * 8 + wave;        // 0..4095
  const int s     = wid >> 9;                    // split 0..7 (uniform per block)
  const int qtile = wid & 511;                   // Q tile 0..511
  const int qbase = qtile * 16;
  const int kbeg  = s * KCHUNK;

  v16bf qf[4];
#pragma unroll
  for (int h4 = 0; h4 < 4; ++h4)
    qf[h4] = load_afrag(q + (size_t)(qbase + n) * HEAD + h4 * 32, hi);

  FragU onesb;                          // row-sum via WMMA (l = P @ 1)
#pragma unroll
  for (int p = 0; p < 8; ++p) onesb.u[p] = 0x3F803F80u;  // bf16 1.0 x2

  v8f O[8] = {};
  v8f L = {};
  float m_r[8];
#pragma unroll
  for (int r = 0; r < 8; ++r) m_r[r] = -__builtin_inff();

  bf16_t* pw = pbuf[wave];

  for (int kbn = kbeg; kbn < kbeg + KCHUNK; kbn += 64) {
    // S = Q K^T for 64 keys (four 16x16 C tiles); base-2 scaled via q
    v8f S[4] = {};
#pragma unroll
    for (int h4 = 0; h4 < 4; ++h4) {
#pragma unroll
      for (int j = 0; j < 4; ++j) {
        const v16bf b =
            load_bfrag(k + (size_t)(kbn + 16 * j + n) * HEAD + h4 * 32, hi);
        S[j] = wmma_bf16(qf[h4], b, S[j]);
      }
    }

    if (kbn + 64 < kbeg + KCHUNK)       // hint next K block toward the caches
      __builtin_prefetch(k + (size_t)(kbn + 64 + lane * 2) * HEAD, 0, 1);

    // online softmax; row (r + 8*hi) lives in one 16-lane half, vgpr r
#pragma unroll
    for (int r = 0; r < 8; ++r) {
      float mloc = fmaxf(fmaxf(S[0][r], S[1][r]), fmaxf(S[2][r], S[3][r]));
      mloc = fmaxf(mloc, __shfl_xor(mloc, 1, 32));
      mloc = fmaxf(mloc, __shfl_xor(mloc, 2, 32));
      mloc = fmaxf(mloc, __shfl_xor(mloc, 4, 32));
      mloc = fmaxf(mloc, __shfl_xor(mloc, 8, 32));
      if (mloc > m_r[r]) {              // lazy rescale: rare after warm-up
        const float scale = exp2f(m_r[r] - mloc);
        m_r[r] = mloc;
        L[r] = L[r] * scale;
#pragma unroll
        for (int t = 0; t < 8; ++t) O[t][r] = O[t][r] * scale;
      }
      const int row = r + 8 * hi;       // C-layout -> row-major P in LDS
#pragma unroll
      for (int j = 0; j < 4; ++j)
        pw[row * 64 + 16 * j + n] = (bf16_t)exp2f(S[j][r] - m_r[r]);
    }

    // reload P as two 16x32 A-fragments (transpose through LDS, same wave)
    const v16bf pf0 = load_afrag((const bf16_t*)pw + n * 64, hi);
    const v16bf pf1 = load_afrag((const bf16_t*)pw + n * 64 + 32, hi);

    L = wmma_bf16(pf0, onesb.v, L);     // row sums
    L = wmma_bf16(pf1, onesb.v, L);

    // O += P V : 8 output column tiles, B from transposed V (contiguous K)
#pragma unroll
    for (int t = 0; t < 8; ++t) {
      const v16bf vf0 = load_bfrag(vT + (size_t)(t * 16 + n) * SEQ + kbn, hi);
      const v16bf vf1 = load_bfrag(vT + (size_t)(t * 16 + n) * SEQ + kbn + 32, hi);
      O[t] = wmma_bf16(pf0, vf0, O[t]);
      O[t] = wmma_bf16(pf1, vf1, O[t]);
    }
  }

  // write split partials: O_p [qtile][s][16][128], m_p/l_p [s][SEQ]
  float* po = pO + (((size_t)qtile * KSPLIT + s) * 16) * HEAD;
#pragma unroll
  for (int r = 0; r < 8; ++r) {
    const int row = r + 8 * hi;
#pragma unroll
    for (int t = 0; t < 8; ++t)
      po[(size_t)row * HEAD + t * 16 + n] = O[t][r];
    if (n == 0) {                       // one lane per half writes row stats
      pm[(size_t)s * SEQ + qbase + row] = m_r[r];
      pl[(size_t)s * SEQ + qbase + row] = L[r];
    }
  }
}

// ---------- Phase 3: combine split-K partials, normalize, store f32
__global__ __launch_bounds__(256) void combine_kernel(
    const float* __restrict__ pO, const float* __restrict__ pm,
    const float* __restrict__ pl, float* __restrict__ out)
{
  const int idx = blockIdx.x * 256 + threadIdx.x;   // 0..SEQ*HEAD-1
  const int row = idx >> 7;
  const int col = idx & 127;
  const int qtile = row >> 4;
  const int r16   = row & 15;

  float m = -__builtin_inff();
#pragma unroll
  for (int s = 0; s < KSPLIT; ++s)
    m = fmaxf(m, pm[(size_t)s * SEQ + row]);

  float acc = 0.0f, l = 0.0f;
#pragma unroll
  for (int s = 0; s < KSPLIT; ++s) {
    const float w = exp2f(pm[(size_t)s * SEQ + row] - m);
    acc += w * pO[(((size_t)qtile * KSPLIT + s) * 16 + r16) * HEAD + col];
    l   += w * pl[(size_t)s * SEQ + row];
  }
  out[idx] = acc / l;
}

extern "C" void kernel_launch(void* const* d_in, const int* in_sizes, int n_in,
                              void* d_out, int out_size, void* d_ws, size_t ws_size,
                              hipStream_t stream) {
  const float* x  = (const float*)d_in[0];
  const float* Wq = (const float*)d_in[1];
  const float* bq = (const float*)d_in[2];
  const float* Wk = (const float*)d_in[3];
  const float* bk = (const float*)d_in[4];
  const float* Wv = (const float*)d_in[5];
  const float* bv = (const float*)d_in[6];
  float* out = (float*)d_out;

  // workspace (bf16): xb 32MB | wtb 1.5MB | q 2MB | k 2MB | vT 2MB  (~39.5MB)
  bf16_t* xb  = (bf16_t*)d_ws;
  bf16_t* wtb = xb  + (size_t)SEQ * EMBED;
  bf16_t* qb  = wtb + (size_t)3 * HEAD * EMBED;
  bf16_t* kb  = qb  + (size_t)SEQ * HEAD;
  bf16_t* vtb = kb  + (size_t)SEQ * HEAD;

  // split-K partials alias dead regions: pO over xb (32MB), pm/pl over wtb
  float* pO = (float*)xb;                          // 512*8*16*128 f32 = 32MB
  float* pm = (float*)wtb;                         // 8*8192 f32 = 256KB
  float* pl = pm + (size_t)KSPLIT * SEQ;           // 8*8192 f32 = 256KB

  pack_kernel<<<8192 + 384, 256, 0, stream>>>(x, Wq, Wk, Wv, xb, wtb);
  proj_kernel<<<1536, 256, 0, stream>>>(xb, wtb, bq, bk, bv, qb, kb, vtb);
  attn_kernel<<<512, 256, 0, stream>>>(qb, kb, vtb, pO, pm, pl);
  combine_kernel<<<(SEQ * HEAD) / 256, 256, 0, stream>>>(pO, pm, pl, out);
}
